// LSTMAttenLSTM_35175782154789
// MI455X (gfx1250) — matrix operand order repaired and checked by reading, not compiled, not run on hardware
//
#include <hip/hip_runtime.h>
#include <stdint.h>
#include <stddef.h>

// ===========================================================================
// 2-layer LSTM encoder + attention LSTM decoder for MI455X (gfx1250).
//
// Strategy:
//  * Batch is fully separable -> persistent workgroups, one 16-row batch tile
//    per WG, whole time loop inside the kernel (no per-step launches).
//  * Recurrent GEMMs ([16 x K] x [K x 2048]) run on v_wmma_f32_16x16x32_bf16,
//    f32 accumulate. Input GEMMs are fused by K-concatenation:
//      layer0: A=[h0|x|0]   K=544,  layer1: A=[h1|h0] K=1024,
//      dec:    A=[h|y|ctx|0] K=1056.
//  * Weights converted once to bf16 and pre-swizzled into the exact WMMA
//    B-fragment lane layout (32B contiguous per lane) -> coalesced b128
//    streaming from L2 (all packed weights ~10.5MB, L2-resident).
//  * B fragments are software-pipelined two tiles deep per wave so each
//    v_wmma waits on loadcnt<=2 instead of loadcnt==0 (round 1 asm showed a
//    full load->wait->wmma serialization per tile).
//  * h kept in LDS as bf16 with bank-conflict-free padded strides; gates in
//    LDS f32 (stride padded) for the elementwise i/f/g/o phase.
//  * Attention (batched matvec, matrix varies per batch row -> not MMA-able)
//    runs one wave per batch row with coalesced bf16 enc reads + shfl_xor
//    reductions + f32 softmax.
// ===========================================================================

typedef __attribute__((ext_vector_type(8)))  float  v8f;
typedef __attribute__((ext_vector_type(8)))  __bf16 v8bf;
typedef __attribute__((ext_vector_type(16))) __bf16 v16bf;

namespace {
constexpr int kB  = 256;
constexpr int kL  = 336;
constexpr int kN  = 8;
constexpr int kH  = 512;
constexpr int kPL = 96;

constexpr int kBT      = 16;            // batch rows per workgroup (= WMMA M)
constexpr int kGN      = 4 * kH;        // 2048 gate columns
constexpr int kNTT     = kGN / 16;      // 128 N-tiles
constexpr int kWaves   = 16;
constexpr int kNTW     = kNTT / kWaves; // 8 N-tiles per wave
constexpr int kThreads = kWaves * 32;   // 512 threads (16 wave32)

constexpr int kKT0 = 17;  // K=544  : 512 h0 + 8 x + 24 zero pad
constexpr int kKT1 = 32;  // K=1024 : 512 h1 + 512 h0
constexpr int kKTD = 33;  // K=1056 : 512 h + 1 y_prev + 512 ctx + 31 zero pad

// LDS strides (halves / floats), padded for bank-conflict-free frag access
constexpr int kA0S = 552;
constexpr int kA1S = 1032;
constexpr int kADS = 1064;
constexpr int kGS  = 2068;

constexpr size_t kEncSmem =
    (size_t)(kBT * kA0S + kBT * kA1S) * 2 +
    (size_t)(kBT * kGS + 2 * kBT * kH) * 4;            // 248,576 B (<320KB)
constexpr size_t kDecSmem =
    (size_t)(kBT * kADS) * 2 +
    (size_t)(kBT * kGS + kBT * kH + kBT * kL + kH + kBT) * 4;  // 222,784 B
}  // namespace

// ----------------------------- helpers ------------------------------------

__device__ __forceinline__ unsigned short f32_to_bf16(float f) {
  union { float f; uint32_t u; } cv; cv.f = f;
  uint32_t u = cv.u;
  if ((u & 0x7F800000u) == 0x7F800000u) {   // inf / nan passthrough
    unsigned short h = (unsigned short)(u >> 16);
    if ((u & 0x007FFFFFu) != 0u) h |= 0x0040;
    return h;
  }
  return (unsigned short)((u + 0x7FFFu + ((u >> 16) & 1u)) >> 16);  // RNE
}

__device__ __forceinline__ float bf16_to_f32(unsigned short h) {
  union { uint32_t u; float f; } cv; cv.u = ((uint32_t)h) << 16;
  return cv.f;
}

__device__ __forceinline__ float sigmoid_f(float x) {
  return 1.0f / (1.0f + __expf(-x));
}

__device__ __forceinline__ float tanh_f(float x) {
  x = fminf(fmaxf(x, -15.0f), 15.0f);
  float e = __expf(2.0f * x);
  return (e - 1.0f) / (e + 1.0f);
}

__device__ __forceinline__ v16bf cat8(v8bf lo, v8bf hi) {
  return __builtin_shufflevector(lo, hi, 0, 1, 2, 3, 4, 5, 6, 7,
                                 8, 9, 10, 11, 12, 13, 14, 15);
}

// A-fragment (16x32 bf16) from LDS row-major h buffer.
// lanes 0-15: M=lane, K {0..15}; lanes 16-31: M=lane-16, K {16..31}
__device__ __forceinline__ v16bf load_a_frag(const unsigned short* Abuf,
                                             int astride, int kt, int lane) {
  const int m  = lane & 15;
  const int kh = (lane >> 4) << 4;  // 0 or 16
  const unsigned short* p = Abuf + m * astride + kt * 32 + kh;
  v8bf lo = *(const v8bf*)(p);        // K base .. base+7   (16B aligned)
  v8bf hi = *(const v8bf*)(p + 8);    // K base+8 .. +15    (16B aligned)
  return cat8(lo, hi);
}

// B-fragment (32x16 bf16) from pre-swizzled weights: 32B contiguous per lane.
__device__ __forceinline__ v16bf load_b_frag(const unsigned short* Wp,
                                             size_t frag, int lane) {
  const unsigned short* p = Wp + frag * 512 + (size_t)lane * 16;
  v8bf lo = *(const v8bf*)(p);
  v8bf hi = *(const v8bf*)(p + 8);
  return cat8(lo, hi);
}

// D = A[16xK] * B[Kx2048] + bias, gates written to LDS (f32, stride kGS).
// B fragments are pipelined 2 tiles deep: each wmma waits on loadcnt<=2.
__device__ __forceinline__ void gemm_gates(const unsigned short* Abuf,
                                           int astride,
                                           const unsigned short* Wp,
                                           const float* __restrict__ bias,
                                           float* gates, int KT, int wave,
                                           int lane) {
  v8f acc[kNTW];
#pragma unroll
  for (int i = 0; i < kNTW; ++i) {
    float bv = bias[(wave * kNTW + i) * 16 + (lane & 15)];
#pragma unroll
    for (int v = 0; v < 8; ++v) acc[i][v] = bv;
  }

  // linear tile index t = kt*8 + i  ->  fragment kt*kNTT + wave*8 + i
  auto fidx = [&](int t) -> size_t {
    return (size_t)(t >> 3) * kNTT + (size_t)(wave * kNTW) + (t & 7);
  };

  v16bf b0 = load_b_frag(Wp, fidx(0), lane);
  v16bf b1 = load_b_frag(Wp, fidx(1), lane);

  for (int kt = 0; kt < KT - 1; ++kt) {
    if (kt + 1 < KT)  // gfx1250 global_prefetch_b8 for next K-slab of weights
      __builtin_prefetch(
          Wp + ((size_t)(kt + 1) * kNTT + wave * kNTW) * 512, 0, 1);
    v16bf a = load_a_frag(Abuf, astride, kt, lane);
#pragma unroll
    for (int i = 0; i < kNTW; ++i) {
      v16bf bn = load_b_frag(Wp, fidx(kt * kNTW + i + 2), lane);
      acc[i] = __builtin_amdgcn_wmma_f32_16x16x32_bf16(
          false, a, false, b0, (short)0, acc[i], false, false);
      b0 = b1;
      b1 = bn;
    }
  }
  {  // peeled last kt: pipeline drain condition is compile-time per i
    v16bf a = load_a_frag(Abuf, astride, KT - 1, lane);
#pragma unroll
    for (int i = 0; i < kNTW; ++i) {
      v16bf bn = b1;
      if (i < kNTW - 2)
        bn = load_b_frag(Wp, fidx((KT - 1) * kNTW + i + 2), lane);
      acc[i] = __builtin_amdgcn_wmma_f32_16x16x32_bf16(
          false, a, false, b0, (short)0, acc[i], false, false);
      b0 = b1;
      b1 = bn;
    }
  }

  const int col0 = lane & 15;
  const int row0 = (lane >> 4) * 8;
#pragma unroll
  for (int i = 0; i < kNTW; ++i) {
    int col = (wave * kNTW + i) * 16 + col0;
#pragma unroll
    for (int v = 0; v < 8; ++v) gates[(row0 + v) * kGS + col] = acc[i][v];
  }
}

__device__ __forceinline__ void load16bf(const unsigned short* p, float* out) {
  const uint4* q = (const uint4*)p;
  uint4 a = q[0];
  uint4 b = q[1];
  uint32_t w[8] = {a.x, a.y, a.z, a.w, b.x, b.y, b.z, b.w};
#pragma unroll
  for (int i = 0; i < 8; ++i) {
    out[2 * i]     = bf16_to_f32((unsigned short)(w[i] & 0xFFFFu));
    out[2 * i + 1] = bf16_to_f32((unsigned short)(w[i] >> 16));
  }
}

// ------------------------- weight prep kernels -----------------------------

// Pack fp32 weights -> bf16 WMMA B-fragments.
// dst[frag*512 + lane*16 + j] = B(k = kt*32 + (lane>>4)*16 + j,
//                                 n = nt*16 + (lane&15)), frag = kt*128+nt.
// mode 0 (layer0, K=544):  k<512 -> W_hh0[n,k]; 512<=k<520 -> W_ih0[n,k-512]
// mode 1 (layer1, K=1024): k<512 -> W_hh1[n,k]; else -> W_ih1[n,k-512]
// mode 2 (dec,   K=1056):  k<512 -> W_dec_hh[n,k]; 512<=k<=1024 ->
//                          W_dec_ih[n, k-512]; else 0
__global__ void pack_weights_kernel(const float* __restrict__ Whh,
                                    const float* __restrict__ Wih,
                                    unsigned short* __restrict__ dst,
                                    int KT, int mode) {
  size_t idx = (size_t)blockIdx.x * blockDim.x + threadIdx.x;
  size_t total = (size_t)KT * kNTT * 512;
  if (idx >= total) return;
  int half = (int)(idx & 511);
  size_t frag = idx >> 9;
  int lane = half >> 4;
  int j = half & 15;
  int kt = (int)(frag / kNTT);
  int nt = (int)(frag % kNTT);
  int n = nt * 16 + (lane & 15);
  int k = kt * 32 + ((lane >> 4) << 4) + j;
  float v = 0.0f;
  if (mode == 0) {
    if (k < kH) v = Whh[(size_t)n * kH + k];
    else if (k < kH + kN) v = Wih[(size_t)n * kN + (k - kH)];
  } else if (mode == 1) {
    if (k < kH) v = Whh[(size_t)n * kH + k];
    else v = Wih[(size_t)n * kH + (k - kH)];
  } else {
    if (k < kH) v = Whh[(size_t)n * kH + k];
    else if (k <= 2 * kH) v = Wih[(size_t)n * (kH + 1) + (k - kH)];
  }
  dst[idx] = f32_to_bf16(v);
}

__global__ void bias_kernel(const float* bi0, const float* bh0,
                            const float* bi1, const float* bh1,
                            const float* bid, const float* bhd,
                            float* o0, float* o1, float* od) {
  int i = blockIdx.x * blockDim.x + threadIdx.x;
  if (i < kGN) {
    o0[i] = bi0[i] + bh0[i];
    o1[i] = bi1[i] + bh1[i];
    od[i] = bid[i] + bhd[i];
  }
}

// --------------------------- encoder kernel --------------------------------

__global__ __launch_bounds__(kThreads) void lstm_encoder_kernel(
    const float* __restrict__ x,
    const unsigned short* __restrict__ WB0,
    const unsigned short* __restrict__ WB1,
    const float* __restrict__ bias0,
    const float* __restrict__ bias1,
    unsigned short* __restrict__ encG,
    unsigned short* __restrict__ hfin,
    float* __restrict__ cfin) {
  extern __shared__ char smemraw[];
  unsigned short* A0 = (unsigned short*)smemraw;          // [16][552] bf16
  unsigned short* A1 = A0 + kBT * kA0S;                   // [16][1032] bf16
  float* gates = (float*)(A1 + kBT * kA1S);               // [16][2068] f32
  float* c0 = gates + kBT * kGS;                          // [16][512]
  float* c1 = c0 + kBT * kH;                              // [16][512]

  const int tid = threadIdx.x;
  const int lane = tid & 31;
  const int wave = tid >> 5;
  const int bg0 = blockIdx.x * kBT;

  for (int i = tid; i < kBT * kA0S; i += kThreads) A0[i] = 0;
  for (int i = tid; i < kBT * kA1S; i += kThreads) A1[i] = 0;
  for (int i = tid; i < kBT * kH; i += kThreads) { c0[i] = 0.f; c1[i] = 0.f; }
  __syncthreads();

  for (int t = 0; t < kL; ++t) {
    // x_t -> A0 cols 512..519
    if (tid < kBT * kN) {
      int b = tid / kN, ii = tid % kN;
      A0[b * kA0S + kH + ii] =
          f32_to_bf16(x[((size_t)(bg0 + b) * kL + t) * kN + ii]);
    }
    __syncthreads();

    gemm_gates(A0, kA0S, WB0, bias0, gates, kKT0, wave, lane);
    __syncthreads();

    // layer0 elementwise: 512 threads x 16 batch rows
    for (int b = 0; b < kBT; ++b) {
      const int j = tid;
      float gi = gates[b * kGS + j];
      float gf = gates[b * kGS + kH + j];
      float gg = gates[b * kGS + 2 * kH + j];
      float go = gates[b * kGS + 3 * kH + j];
      float cc = sigmoid_f(gf) * c0[b * kH + j] + sigmoid_f(gi) * tanh_f(gg);
      float h = sigmoid_f(go) * tanh_f(cc);
      c0[b * kH + j] = cc;
      unsigned short hb = f32_to_bf16(h);
      A0[b * kA0S + j] = hb;       // h0 for next step's layer0
      A1[b * kA1S + kH + j] = hb;  // layer1 input
    }
    __syncthreads();

    gemm_gates(A1, kA1S, WB1, bias1, gates, kKT1, wave, lane);
    __syncthreads();

    // layer1 elementwise + encoder output store
    for (int b = 0; b < kBT; ++b) {
      const int j = tid;
      float gi = gates[b * kGS + j];
      float gf = gates[b * kGS + kH + j];
      float gg = gates[b * kGS + 2 * kH + j];
      float go = gates[b * kGS + 3 * kH + j];
      float cc = sigmoid_f(gf) * c1[b * kH + j] + sigmoid_f(gi) * tanh_f(gg);
      float h = sigmoid_f(go) * tanh_f(cc);
      c1[b * kH + j] = cc;
      unsigned short hb = f32_to_bf16(h);
      A1[b * kA1S + j] = hb;
      encG[((size_t)(bg0 + b) * kL + t) * kH + j] = hb;
      if (t == kL - 1) {
        hfin[(size_t)(bg0 + b) * kH + j] = hb;
        cfin[(size_t)(bg0 + b) * kH + j] = cc;
      }
    }
    __syncthreads();
  }
}

// --------------------------- decoder kernel --------------------------------

__global__ __launch_bounds__(kThreads) void lstm_decoder_kernel(
    const unsigned short* __restrict__ encG,
    const unsigned short* __restrict__ hfin,
    const float* __restrict__ cfin,
    const unsigned short* __restrict__ WBD,
    const float* __restrict__ biasd,
    const float* __restrict__ Wfc,
    const float* __restrict__ bfc,
    float* __restrict__ out) {
  extern __shared__ char smemraw[];
  unsigned short* A = (unsigned short*)smemraw;  // [16][1064] bf16 (h|y|ctx|0)
  float* gates = (float*)(A + kBT * kADS);       // [16][2068]
  float* c = gates + kBT * kGS;                  // [16][512]
  float* wbuf = c + kBT * kH;                    // [16][336] attn weights
  float* wfcS = wbuf + kBT * kL;                 // [512]
  float* yacc = wfcS + kH;                       // [16]

  const int tid = threadIdx.x;
  const int lane = tid & 31;
  const int wave = tid >> 5;
  const int bg0 = blockIdx.x * kBT;

  for (int i = tid; i < kBT * kADS; i += kThreads) A[i] = 0;
  __syncthreads();
  for (int b = 0; b < kBT; ++b) {
    A[b * kADS + tid] = hfin[(size_t)(bg0 + b) * kH + tid];
    c[b * kH + tid] = cfin[(size_t)(bg0 + b) * kH + tid];
  }
  wfcS[tid] = Wfc[tid];
  __syncthreads();

  const float bfc0 = bfc[0];

  for (int s = 0; s < kPL; ++s) {
    // ---------- attention: one wave per batch row ----------
    {
      const int b = wave;
      const unsigned short* encp = encG + (size_t)(bg0 + b) * kL * kH;
      float hreg[16];
#pragma unroll
      for (int q = 0; q < 16; ++q)
        hreg[q] = bf16_to_f32(A[b * kADS + lane * 16 + q]);

      for (int l = 0; l < kL; ++l) {
        float e[16];
        load16bf(encp + (size_t)l * kH + lane * 16, e);
        float sc = 0.f;
#pragma unroll
        for (int q = 0; q < 16; ++q) sc = fmaf(e[q], hreg[q], sc);
#pragma unroll
        for (int off = 16; off > 0; off >>= 1) sc += __shfl_xor(sc, off, 32);
        if (lane == 0) wbuf[b * kL + l] = sc;
      }
      __builtin_amdgcn_wave_barrier();

      float mx = -3.0e38f;
      for (int l = lane; l < kL; l += 32) mx = fmaxf(mx, wbuf[b * kL + l]);
#pragma unroll
      for (int off = 16; off > 0; off >>= 1)
        mx = fmaxf(mx, __shfl_xor(mx, off, 32));
      float sum = 0.f;
      for (int l = lane; l < kL; l += 32) {
        float e = __expf(wbuf[b * kL + l] - mx);
        wbuf[b * kL + l] = e;
        sum += e;
      }
#pragma unroll
      for (int off = 16; off > 0; off >>= 1) sum += __shfl_xor(sum, off, 32);
      const float inv = 1.0f / sum;
      __builtin_amdgcn_wave_barrier();

      float ctx[16];
#pragma unroll
      for (int q = 0; q < 16; ++q) ctx[q] = 0.f;
      for (int l = 0; l < kL; ++l) {
        float wl = wbuf[b * kL + l] * inv;
        float e[16];
        load16bf(encp + (size_t)l * kH + lane * 16, e);
#pragma unroll
        for (int q = 0; q < 16; ++q) ctx[q] = fmaf(wl, e[q], ctx[q]);
      }
#pragma unroll
      for (int q = 0; q < 16; ++q)
        A[b * kADS + kH + 1 + lane * 16 + q] = f32_to_bf16(ctx[q]);
    }
    if (tid < kBT) yacc[tid] = 0.f;
    __syncthreads();

    gemm_gates(A, kADS, WBD, biasd, gates, kKTD, wave, lane);
    __syncthreads();

    for (int b = 0; b < kBT; ++b) {
      const int j = tid;
      float gi = gates[b * kGS + j];
      float gf = gates[b * kGS + kH + j];
      float gg = gates[b * kGS + 2 * kH + j];
      float go = gates[b * kGS + 3 * kH + j];
      float cc = sigmoid_f(gf) * c[b * kH + j] + sigmoid_f(gi) * tanh_f(gg);
      float h = sigmoid_f(go) * tanh_f(cc);
      c[b * kH + j] = cc;
      A[b * kADS + j] = f32_to_bf16(h);
      atomicAdd(&yacc[b], h * wfcS[j]);  // ds_add_f32 reduction for FC
    }
    __syncthreads();

    if (tid < kBT) {
      float y = yacc[tid] + bfc0;
      out[(size_t)(bg0 + tid) * kPL + s] = y;
      A[tid * kADS + kH] = f32_to_bf16(y);  // y_prev for next step
    }
    __syncthreads();
  }
}

// ------------------------------- host side ---------------------------------

extern "C" void kernel_launch(void* const* d_in, const int* in_sizes, int n_in,
                              void* d_out, int out_size, void* d_ws,
                              size_t ws_size, hipStream_t stream) {
  (void)in_sizes; (void)n_in; (void)out_size; (void)ws_size;

  const float* x        = (const float*)d_in[0];
  const float* W_ih0    = (const float*)d_in[1];
  const float* W_hh0    = (const float*)d_in[2];
  const float* b_ih0    = (const float*)d_in[3];
  const float* b_hh0    = (const float*)d_in[4];
  const float* W_ih1    = (const float*)d_in[5];
  const float* W_hh1    = (const float*)d_in[6];
  const float* b_ih1    = (const float*)d_in[7];
  const float* b_hh1    = (const float*)d_in[8];
  const float* W_dec_ih = (const float*)d_in[9];
  const float* W_dec_hh = (const float*)d_in[10];
  const float* b_dec_ih = (const float*)d_in[11];
  const float* b_dec_hh = (const float*)d_in[12];
  const float* W_fc     = (const float*)d_in[13];
  const float* b_fc     = (const float*)d_in[14];

  // workspace carve (~100MB): packed weights + biases + enc_outs + h/c final
  char* ws = (char*)d_ws;
  size_t off = 0;
  auto take = [&](size_t bytes) -> char* {
    char* p = ws + off;
    off += (bytes + 255) & ~(size_t)255;
    return p;
  };
  unsigned short* WB0 = (unsigned short*)take((size_t)kKT0 * kNTT * 512 * 2);
  unsigned short* WB1 = (unsigned short*)take((size_t)kKT1 * kNTT * 512 * 2);
  unsigned short* WBD = (unsigned short*)take((size_t)kKTD * kNTT * 512 * 2);
  float* bias0 = (float*)take((size_t)kGN * 4);
  float* bias1 = (float*)take((size_t)kGN * 4);
  float* biasd = (float*)take((size_t)kGN * 4);
  unsigned short* encG = (unsigned short*)take((size_t)kB * kL * kH * 2);
  unsigned short* hfin = (unsigned short*)take((size_t)kB * kH * 2);
  float* cfin = (float*)take((size_t)kB * kH * 4);

  {
    size_t tot0 = (size_t)kKT0 * kNTT * 512;
    pack_weights_kernel<<<dim3((unsigned)((tot0 + 255) / 256)), dim3(256), 0,
                          stream>>>(W_hh0, W_ih0, WB0, kKT0, 0);
    size_t tot1 = (size_t)kKT1 * kNTT * 512;
    pack_weights_kernel<<<dim3((unsigned)((tot1 + 255) / 256)), dim3(256), 0,
                          stream>>>(W_hh1, W_ih1, WB1, kKT1, 1);
    size_t totd = (size_t)kKTD * kNTT * 512;
    pack_weights_kernel<<<dim3((unsigned)((totd + 255) / 256)), dim3(256), 0,
                          stream>>>(W_dec_hh, W_dec_ih, WBD, kKTD, 2);
    bias_kernel<<<dim3((kGN + 255) / 256), dim3(256), 0, stream>>>(
        b_ih0, b_hh0, b_ih1, b_hh1, b_dec_ih, b_dec_hh, bias0, bias1, biasd);
  }

  lstm_encoder_kernel<<<dim3(kB / kBT), dim3(kThreads), kEncSmem, stream>>>(
      x, WB0, WB1, bias0, bias1, encG, hfin, cfin);

  lstm_decoder_kernel<<<dim3(kB / kBT), dim3(kThreads), kDecSmem, stream>>>(
      encG, hfin, cfin, WBD, biasd, W_fc, b_fc, (float*)d_out);
}
